// Weight_Distributer_decoupled_53523882443611
// MI455X (gfx1250) — matrix operand order repaired and checked by reading, not compiled, not run on hardware
//
#include <hip/hip_runtime.h>

// ---------------------------------------------------------------------------
// Types for WMMA fragments (gfx1250, wave32)
// ---------------------------------------------------------------------------
typedef _Float16 v16h __attribute__((ext_vector_type(16)));
typedef _Float16 v8h  __attribute__((ext_vector_type(8)));
typedef float    v8f  __attribute__((ext_vector_type(8)));

#define NCLASS 64
#define TOPK   16
#define NHID   256
#define DEGREE 9
#define H1DIM  512   // 2*NHID
#define MLP_WAVES 4  // waves per block in MLP kernel

#define LOG2E 1.44269504088896340736f
#define LN2   0.69314718055994530942f

// Branchless native transcendentals (single v_exp_f32 / v_log_f32 / v_rcp_f32)
__device__ __forceinline__ float fast_exp(float x) {
  return __builtin_amdgcn_exp2f(x * LOG2E);
}
__device__ __forceinline__ float fast_log(float x) {
  return __builtin_amdgcn_logf(x) * LN2;
}
// tanh(x) = 1 - 2/(e^{2x}+1); exp2 saturates to +inf/0 => rcp gives exact +/-1
__device__ __forceinline__ float fast_tanh(float x) {
  float e = __builtin_amdgcn_exp2f(x * (2.0f * LOG2E));
  return 1.0f - 2.0f * __builtin_amdgcn_rcpf(e + 1.0f);
}

// ---------------------------------------------------------------------------
// Kernel 1: weight prep — convert W1/W2/W3 to f16, pre-swizzled into the
// WMMA B-fragment layout (ISA 7.12.2): for a 32(K)x16(N) f16 B tile,
// lane n in [0,16) holds K=0..7,16..23 of column n, lane n+16 holds K=8..15,24..31.
// Each tile stored as 32 lanes x 16 halves, contiguous per lane.
// W1 is K-padded 16->32 with zeros; W3 is N-padded 9->16 with zeros.
// ---------------------------------------------------------------------------
__global__ void prep_kernel(const float* __restrict__ W1,
                            const float* __restrict__ W2,
                            const float* __restrict__ W3,
                            _Float16* __restrict__ W1s,
                            _Float16* __restrict__ W2s,
                            _Float16* __restrict__ W3s) {
  const int W1S = 32 * 32 * 16;   // 32 N-tiles
  const int W2S = 256 * 32 * 16;  // 16 K-tiles x 16 N-tiles
  const int W3S = 8 * 32 * 16;    // 8 K-tiles x 1 N-tile
  int idx = blockIdx.x * blockDim.x + threadIdx.x;
  if (idx < W1S) {
    int nt  = idx >> 9;          // tile index (512 halves per tile)
    int rem = idx & 511;
    int L = rem >> 4, h = rem & 15;
    int n = nt * 16 + (L & 15);
    int k = ((L < 16) ? 0 : 8) + ((h < 8) ? 0 : 8) + h;
    float v = (k < TOPK) ? W1[k * H1DIM + n] : 0.0f;
    W1s[idx] = (_Float16)v;
  } else if (idx < W1S + W2S) {
    int i2  = idx - W1S;
    int tile = i2 >> 9;
    int rem  = i2 & 511;
    int L = rem >> 4, h = rem & 15;
    int kt = tile >> 4, nt = tile & 15;
    int k = kt * 32 + ((L < 16) ? 0 : 8) + ((h < 8) ? 0 : 8) + h;
    int n = nt * 16 + (L & 15);
    W2s[i2] = (_Float16)W2[k * NHID + n];
  } else if (idx < W1S + W2S + W3S) {
    int i3  = idx - W1S - W2S;
    int kt  = i3 >> 9;
    int rem = i3 & 511;
    int L = rem >> 4, h = rem & 15;
    int k = kt * 32 + ((L < 16) ? 0 : 8) + ((h < 8) ? 0 : 8) + h;
    int n = L & 15;
    float v = (n < DEGREE) ? W3[k * DEGREE + n] : 0.0f;
    W3s[i3] = (_Float16)v;
  }
}

// ---------------------------------------------------------------------------
// Kernel 2: softmax over 64 classes + sorted top-16 -> f16 [N,32] (K-padded).
// One wave32 per node; 2 classes per lane; shuffle reductions.
// ---------------------------------------------------------------------------
__global__ void topk_kernel(const float* __restrict__ g0,
                            _Float16* __restrict__ rankedh, int n) {
  int lane = threadIdx.x & 31;
  int node = (blockIdx.x * blockDim.x + threadIdx.x) >> 5;
  if (node >= n) return;
  float v0 = g0[(size_t)node * NCLASS + lane];
  float v1 = g0[(size_t)node * NCLASS + 32 + lane];
  float m = fmaxf(v0, v1);
  for (int o = 16; o; o >>= 1) m = fmaxf(m, __shfl_xor(m, o, 32));
  float e0 = fast_exp(v0 - m), e1 = fast_exp(v1 - m);
  float s = e0 + e1;
  for (int o = 16; o; o >>= 1) s += __shfl_xor(s, o, 32);
  float inv = __builtin_amdgcn_rcpf(s);
  float c0 = e0 * inv, c1 = e1 * inv;  // probs in (0,1]
  float keep = 0.0f;
  for (int j = 0; j < TOPK; ++j) {
    float lv; int li;
    if (c0 >= c1) { lv = c0; li = lane * 2; } else { lv = c1; li = lane * 2 + 1; }
    for (int o = 16; o; o >>= 1) {
      float ov = __shfl_xor(lv, o, 32);
      int   oi = __shfl_xor(li, o, 32);
      if (ov > lv || (ov == lv && oi < li)) { lv = ov; li = oi; }
    }
    if ((li >> 1) == lane) { if (li & 1) c1 = -1.0f; else c0 = -1.0f; }
    if (lane == j) keep = lv;
  }
  // lanes 0..15 hold ranked[0..15]; lanes 16..31 write the zero K-padding
  rankedh[(size_t)node * 32 + lane] = (lane < TOPK) ? (_Float16)keep : (_Float16)0.0f;
}

// ---------------------------------------------------------------------------
// Kernel 3: fused 3-layer MLP with v_wmma_f32_16x16x32_f16.
// One wave per 16-node tile; 4 waves/block; h1/h2 staged in LDS as f16.
// A-fragment (16x32 f16): lanes 0-15 = rows M, halves 0..7 -> K0..7,
// halves 8..15 -> K16..23; lanes 16-31 same rows, K8..15 / K24..31.
// GEMM2 holds 8 independent accumulators so each A fragment is loaded once
// per K-step and the 8 WMMAs pipeline without D->C serialization.
// ---------------------------------------------------------------------------
__device__ __forceinline__ v16h cat8(v8h lo, v8h hi) {
  v16h r;
#pragma unroll
  for (int i = 0; i < 8; ++i) { r[i] = lo[i]; r[i + 8] = hi[i]; }
  return r;
}

__global__ void mlp_kernel(const _Float16* __restrict__ rankedh,
                           const _Float16* __restrict__ W1s,
                           const _Float16* __restrict__ W2s,
                           const _Float16* __restrict__ W3s,
                           const float* __restrict__ b1,
                           const float* __restrict__ b2,
                           const float* __restrict__ b3,
                           float* __restrict__ out, int n_nodes) {
  extern __shared__ _Float16 lds[];
  const int lane = threadIdx.x & 31;
  const int wave = threadIdx.x >> 5;
  const int lrow = lane & 15;   // row (A) or column (B/C) within tile
  const int hi   = lane >> 4;   // which K/M half this lane covers
  _Float16* h1 = lds + (size_t)wave * (16 * H1DIM);
  _Float16* h2 = lds + (size_t)MLP_WAVES * 16 * H1DIM + (size_t)wave * (16 * NHID);
  const int m0 = (blockIdx.x * MLP_WAVES + wave) * 16;
  int row  = m0 + lrow;
  int rowc = (row < n_nodes) ? row : (n_nodes - 1);

  // ---- load A fragment of ranked probs (16x32, zeros in K>=16) ----
  v8h a0 = *(const v8h*)(rankedh + (size_t)rowc * 32 + hi * 8);
  v8h a1 = *(const v8h*)(rankedh + (size_t)rowc * 32 + 16 + hi * 8);
  v16h arank = cat8(a0, a1);

  // ---- GEMM1: [16,32] x [32,512] -> tanh -> h1 (LDS, f16 row-major) ----
  for (int nt = 0; nt < 32; ++nt) {
    const v8h* q = (const v8h*)(W1s + ((size_t)nt * 32 + lane) * 16);
    v16h b = cat8(q[0], q[1]);
    v8f acc = {};
    acc = __builtin_amdgcn_wmma_f32_16x16x32_f16(false, arank, false, b,
                                                 (short)0, acc, false, false);
    float bias = b1[nt * 16 + lrow];
#pragma unroll
    for (int r = 0; r < 8; ++r) {
      int mm = hi ? (8 + r) : r;
      h1[mm * H1DIM + nt * 16 + lrow] = (_Float16)fast_tanh(acc[r] + bias);
    }
  }
  __syncthreads();

  // ---- GEMM2: [16,512] x [512,256] -> tanh -> h2 ----
  // 2 groups of 8 N-tiles; A loaded once per K-step per group.
  for (int ng = 0; ng < 2; ++ng) {
    v8f acc[8];
#pragma unroll
    for (int j = 0; j < 8; ++j) acc[j] = (v8f){};
    for (int kt = 0; kt < 16; ++kt) {
      v8h pa0 = *(const v8h*)(h1 + lrow * H1DIM + kt * 32 + hi * 8);
      v8h pa1 = *(const v8h*)(h1 + lrow * H1DIM + kt * 32 + 16 + hi * 8);
      v16h a = cat8(pa0, pa1);
#pragma unroll
      for (int j = 0; j < 8; ++j) {
        int nt = ng * 8 + j;
        const v8h* q = (const v8h*)(W2s + (((size_t)kt * 16 + nt) * 32 + lane) * 16);
        v16h b = cat8(q[0], q[1]);
        acc[j] = __builtin_amdgcn_wmma_f32_16x16x32_f16(false, a, false, b,
                                                        (short)0, acc[j], false, false);
      }
    }
#pragma unroll
    for (int j = 0; j < 8; ++j) {
      int nt = ng * 8 + j;
      float bias = b2[nt * 16 + lrow];
#pragma unroll
      for (int r = 0; r < 8; ++r) {
        int mm = hi ? (8 + r) : r;
        h2[mm * NHID + nt * 16 + lrow] = (_Float16)fast_tanh(acc[j][r] + bias);
      }
    }
  }
  __syncthreads();

  // ---- GEMM3: [16,256] x [256,16(pad of 9)] -> weight ----
  v8f acc = {};
  for (int kt = 0; kt < 8; ++kt) {
    v8h pa0 = *(const v8h*)(h2 + lrow * NHID + kt * 32 + hi * 8);
    v8h pa1 = *(const v8h*)(h2 + lrow * NHID + kt * 32 + 16 + hi * 8);
    v16h a = cat8(pa0, pa1);
    const v8h* q = (const v8h*)(W3s + ((size_t)kt * 32 + lane) * 16);
    v16h b = cat8(q[0], q[1]);
    acc = __builtin_amdgcn_wmma_f32_16x16x32_f16(false, a, false, b,
                                                 (short)0, acc, false, false);
  }
  int n = lrow;
  float b3v = (n < DEGREE) ? b3[n] : 0.0f;
  float* wgt = out + (size_t)n_nodes * NCLASS;
#pragma unroll
  for (int r = 0; r < 8; ++r) {
    int mm = hi ? (8 + r) : r;
    int node = m0 + mm;
    if (node < n_nodes && n < DEGREE)
      wgt[(size_t)node * DEGREE + n] = acc[r] + b3v;
  }
}

// ---------------------------------------------------------------------------
// Kernel 4: SPMM  dst[rows[e]] += vals[e] * src[cols[e]]   (64 f32 per row)
// 2 edges per wave: 16 lanes x float4 covers one 64-class row.
// ---------------------------------------------------------------------------
__global__ void spmm_kernel(const int* __restrict__ rows,
                            const int* __restrict__ cols,
                            const float* __restrict__ vals,
                            const float* __restrict__ src,
                            float* __restrict__ dst, int nE) {
  int tid  = blockIdx.x * blockDim.x + threadIdx.x;
  int lane = tid & 31;
  int e    = (tid >> 5) * 2 + (lane >> 4);
  if (e >= nE) return;
  int   c  = cols[e];
  int   rr = rows[e];
  float v  = vals[e];
  int   c0 = (lane & 15) * 4;
  float4 f = *(const float4*)(src + (size_t)c * NCLASS + c0);
  float* d = dst + (size_t)rr * NCLASS + c0;
  unsafeAtomicAdd(d + 0, v * f.x);
  unsafeAtomicAdd(d + 1, v * f.y);
  unsafeAtomicAdd(d + 2, v * f.z);
  unsafeAtomicAdd(d + 3, v * f.w);
}

// ---------------------------------------------------------------------------
// Kernel 5: final = (accum ? final : 0) + wgt[:,widx] * f
// ---------------------------------------------------------------------------
__global__ void axpy_kernel(float* __restrict__ final_, const float* __restrict__ f,
                            const float* __restrict__ wgt, int widx, int accum, int n) {
  int tid  = blockIdx.x * blockDim.x + threadIdx.x;
  int node = tid >> 4;
  if (node >= n) return;
  int c0   = (tid & 15) * 4;
  float w  = wgt[(size_t)node * DEGREE + widx];
  float4 fv = *(const float4*)(f + (size_t)node * NCLASS + c0);
  float4* dp = (float4*)(final_ + (size_t)node * NCLASS + c0);
  float4 r;
  if (accum) {
    float4 cur = *dp;
    r = make_float4(cur.x + w * fv.x, cur.y + w * fv.y,
                    cur.z + w * fv.z, cur.w + w * fv.w);
  } else {
    r = make_float4(w * fv.x, w * fv.y, w * fv.z, w * fv.w);
  }
  *dp = r;
}

// ---------------------------------------------------------------------------
// Kernel 6: in-place log_softmax over 64 classes, one wave per node.
// ---------------------------------------------------------------------------
__global__ void logsoftmax_kernel(float* __restrict__ final_, int n) {
  int lane = threadIdx.x & 31;
  int node = (blockIdx.x * blockDim.x + threadIdx.x) >> 5;
  if (node >= n) return;
  float x0 = final_[(size_t)node * NCLASS + lane];
  float x1 = final_[(size_t)node * NCLASS + 32 + lane];
  float m = fmaxf(x0, x1);
  for (int o = 16; o; o >>= 1) m = fmaxf(m, __shfl_xor(m, o, 32));
  float s = fast_exp(x0 - m) + fast_exp(x1 - m);
  for (int o = 16; o; o >>= 1) s += __shfl_xor(s, o, 32);
  float ls = fast_log(s);
  final_[(size_t)node * NCLASS + lane]      = x0 - m - ls;
  final_[(size_t)node * NCLASS + 32 + lane] = x1 - m - ls;
}

// ---------------------------------------------------------------------------
// Host launcher
// ---------------------------------------------------------------------------
extern "C" void kernel_launch(void* const* d_in, const int* in_sizes, int n_in,
                              void* d_out, int out_size, void* d_ws, size_t ws_size,
                              hipStream_t stream) {
  const float* f_pred = (const float*)d_in[0];
  const float* g0     = (const float*)d_in[1];
  const int*   rows   = (const int*)d_in[2];
  const int*   cols   = (const int*)d_in[3];
  const float* vals   = (const float*)d_in[4];
  const float* W1     = (const float*)d_in[5];
  const float* b1     = (const float*)d_in[6];
  const float* W2     = (const float*)d_in[7];
  const float* b2     = (const float*)d_in[8];
  const float* W3     = (const float*)d_in[9];
  const float* b3     = (const float*)d_in[10];
  const int N = in_sizes[0] / NCLASS;
  const int E = in_sizes[2];

  float* out = (float*)d_out;
  float* wgt = out + (size_t)N * NCLASS;  // [N, DEGREE] output #2

  // workspace carve-up (256B-aligned pieces)
  char* ws = (char*)d_ws;
  size_t off = 0;
  _Float16* rankedh = (_Float16*)(ws + off); off += (size_t)N * 32 * 2;
  off = (off + 255) & ~(size_t)255;
  _Float16* W1s = (_Float16*)(ws + off); off += (size_t)32 * 32 * 16 * 2;
  _Float16* W2s = (_Float16*)(ws + off); off += (size_t)256 * 32 * 16 * 2;
  _Float16* W3s = (_Float16*)(ws + off); off += (size_t)8 * 32 * 16 * 2;
  off = (off + 255) & ~(size_t)255;
  float* fA = (float*)(ws + off); off += (size_t)N * NCLASS * 4;
  float* fB = (float*)(ws + off);

  // 1) weight prep (f16 + B-fragment swizzle)
  prep_kernel<<<(16384 + 131072 + 4096) / 256, 256, 0, stream>>>(W1, W2, W3, W1s, W2s, W3s);

  // 2) softmax + top-16
  topk_kernel<<<(N * 32 + 255) / 256, 256, 0, stream>>>(g0, rankedh, N);

  // 3) WMMA MLP -> weight[N,9] into d_out
  {
    int blocks = (N + 16 * MLP_WAVES - 1) / (16 * MLP_WAVES);
    size_t shmem = (size_t)MLP_WAVES * (16 * H1DIM + 16 * NHID) * sizeof(_Float16); // 96KB
    mlp_kernel<<<blocks, MLP_WAVES * 32, shmem, stream>>>(rankedh, W1s, W2s, W3s,
                                                          b1, b2, b3, out, N);
  }

  // 4) final = w0 * f_pred ; then 8 rounds of SPMM + axpy
  int axpyBlocks = (N * 16 + 255) / 256;
  axpy_kernel<<<axpyBlocks, 256, 0, stream>>>(out, f_pred, wgt, 0, 0, N);

  const float* src = f_pred;
  for (int i = 0; i < DEGREE - 1; ++i) {
    float* dst = (i & 1) ? fB : fA;
    hipMemsetAsync(dst, 0, (size_t)N * NCLASS * sizeof(float), stream);
    long long waves = (E + 1) / 2;
    long long threads = waves * 32;
    int blocks = (int)((threads + 255) / 256);
    spmm_kernel<<<blocks, 256, 0, stream>>>(rows, cols, vals, src, dst, E);
    axpy_kernel<<<axpyBlocks, 256, 0, stream>>>(out, dst, wgt, i + 1, 1, N);
    src = dst;
  }

  // 5) in-place log_softmax on the first N*64 floats of d_out
  logsoftmax_kernel<<<(N * 32 + 255) / 256, 256, 0, stream>>>(out, N);
}
